// SymmetricContraction_27908697489828
// MI455X (gfx1250) — matrix operand order repaired and checked by reading, not compiled, not run on hardware
//
#include <hip/hip_runtime.h>

// ---------------------------------------------------------------------------
// Symmetric contraction (MACE-style), MI455X / gfx1250.
//
// result[b,c] = sum_{w,x,i,k} U3[w,x,i,k] W3[b,k,c] x_w x_x x_i
//             + sum_{w,x,k}   U2[w,x,k]   W2[b,k,c] x_w x_x
//             + sum_{w,k}     U1[w,k]     W1[b,k,c] x_w
// with W*[b,k,c] = sum_e y[b,e] w*[e,k,c].
//
// U3/U2 are symmetrized onto sorted multi-indices -> 816 triples + 136 pairs
// + 16 singles = 968 monomial rows, padded to K=1024, N=32 output columns
// (23 U3-k | 4 U2-k | 1 U1 | 4 pad).  Per node: one f16 WMMA GEMM
// S[32x128] = Usym^T[32x1024] x Mono[1024x128], then dot with Wm[n,c].
// ---------------------------------------------------------------------------

#define CCH   128
#define LL    16
#define NP3   23
#define NP2   4
#define NE    10
#define KK    1024
#define R_TRI 816
#define R_PAI 952   // 816 + 136
#define R_SIN 968   // + 16

typedef __attribute__((ext_vector_type(16))) _Float16 v16h;
typedef __attribute__((ext_vector_type(8)))  float    v8f;

// Lexicographic enumeration of sorted triples/pairs/singles over 16 values.
// a,b,c index an extended register file: xv[0..15]=x, xv[16]=1.0, xv[17]=0.0
struct RowTab {
  unsigned char a[KK], b[KK], c[KK];
  constexpr RowTab() : a(), b(), c() {
    int r = 0;
    for (int i = 0; i < 16; ++i)
      for (int j = i; j < 16; ++j)
        for (int k = j; k < 16; ++k) { a[r]=(unsigned char)i; b[r]=(unsigned char)j; c[r]=(unsigned char)k; ++r; }
    for (int i = 0; i < 16; ++i)
      for (int j = i; j < 16; ++j)   { a[r]=(unsigned char)i; b[r]=(unsigned char)j; c[r]=16; ++r; }
    for (int i = 0; i < 16; ++i)     { a[r]=(unsigned char)i; b[r]=16; c[r]=16; ++r; }
    for (; r < KK; ++r)              { a[r]=17; b[r]=17; c[r]=17; }
  }
};
static constexpr RowTab ROWS{};

// Runtime decode (prep kernel) -- must match RowTab enumeration order.
__device__ inline void decode_row(int r, int& a, int& b, int& c) {
  if (r < R_TRI) {
    int t = r;
    int aa = 0, bb = 0;
    for (aa = 0; aa < 16; ++aa) { int cnt = (16 - aa) * (17 - aa) / 2; if (t < cnt) break; t -= cnt; }
    for (bb = aa; bb < 16; ++bb) { int cnt = 16 - bb; if (t < cnt) break; t -= cnt; }
    a = aa; b = bb; c = bb + t;
  } else if (r < R_PAI) {
    int t = r - R_TRI;
    int aa = 0;
    for (aa = 0; aa < 16; ++aa) { int cnt = 16 - aa; if (t < cnt) break; t -= cnt; }
    a = aa; b = aa + t; c = 16;
  } else if (r < R_SIN) {
    a = r - R_PAI; b = 16; c = 16;
  } else {
    a = 17; b = 17; c = 17;
  }
}

__device__ inline float sym_u3(const float* __restrict__ U3, int a, int b, int c, int k) {
  // Sum U3 over all DISTINCT permutations of (a,b,c) at path k.
  int P[6][3] = { {a,b,c},{a,c,b},{b,a,c},{b,c,a},{c,a,b},{c,b,a} };
  float s = 0.f;
  #pragma unroll
  for (int i = 0; i < 6; ++i) {
    bool dup = false;
    for (int j = 0; j < i; ++j)
      if (P[j][0]==P[i][0] && P[j][1]==P[i][1] && P[j][2]==P[i][2]) dup = true;
    if (!dup) s += U3[((P[i][0]*LL + P[i][1])*LL + P[i][2])*NP3 + k];
  }
  return s;
}

// ---------------------------------------------------------------------------
// Prep: build the pre-swizzled f16 A-operand (Usym^T, M=32 x K=1024) in d_ws.
// Swizzle follows the ISA 16-bit A-matrix 16x32 layout:
//   lane l (h=l/16, m=l%16) holds row M=m; VGPR0-3: K = 32kb + 8h + {0..7},
//   VGPR4-7: K = 32kb + 16 + 8h + {0..7}.  Two 16x1024 M-tiles back to back.
// Deterministic gather form (no atomics).
// ---------------------------------------------------------------------------
__global__ __launch_bounds__(256) void symcon_prep_A(
    const float* __restrict__ U3, const float* __restrict__ U2,
    const float* __restrict__ U1, _Float16* __restrict__ wsA) {
  int t = blockIdx.x * blockDim.x + threadIdx.x;   // 32768 elements
  if (t >= 2 * 32 * 32 * 16) return;
  int j    = t & 15;
  int lane = (t >> 4) & 31;
  int kb   = (t >> 9) & 31;
  int Mt   = t >> 14;
  int h = lane >> 4, ln = lane & 15;
  int n = Mt * 16 + ln;                                   // output column
  int Kidx = kb * 32 + ((j < 8) ? (h * 8 + j) : (16 + h * 8 + (j - 8)));
  int a, b, c;
  decode_row(Kidx, a, b, c);
  float v = 0.f;
  if (Kidx < R_TRI) {
    if (n < NP3) v = sym_u3(U3, a, b, c, n);
  } else if (Kidx < R_PAI) {
    if (n >= NP3 && n < NP3 + NP2) {
      int k = n - NP3;
      v = U2[(a*LL + b)*NP2 + k] + ((a != b) ? U2[(b*LL + a)*NP2 + k] : 0.f);
    }
  } else if (Kidx < R_SIN) {
    if (n == NP3 + NP2) v = U1[Kidx - R_PAI];
  }
  wsA[t] = (_Float16)v;
}

// ---------------------------------------------------------------------------
// Main: one block (256 threads = 8 waves) per node b.
// Wave w handles channels [16w, 16w+16): builds monomial B-fragments in
// registers (compile-time triple tables, cndmask half-select -> no branch,
// EXEC stays all-ones for WMMA), accumulates 2 M-tiles via
// v_wmma_f32_16x16x32_f16, then dots f32 accumulators with Wm.
// B-fragment layout (matches A table order):
//   lane l holds N = l%16, K = 8*(l/16) + j (j<8), 16 + 8*(l/16) + (j-8) (j>=8), packed 2 K per VGPR.
// ---------------------------------------------------------------------------
__global__ __launch_bounds__(256) void symcon_main(
    const float* __restrict__ x,  const float* __restrict__ y,
    const float* __restrict__ w3, const float* __restrict__ w2,
    const float* __restrict__ w1, const _Float16* __restrict__ wsA,
    float* __restrict__ out) {
  __shared__ float Wm[32 * CCH];   // element-mixed weights, rows 28..31 = 0
  __shared__ float yl[16];

  const int b   = blockIdx.x;
  const int tid = threadIdx.x;

  if (tid < NE) yl[tid] = y[b * NE + tid];
  __syncthreads();

  // Wm[n][ch] = sum_e y[b,e] * w(e,n,ch)   (4096 entries over 256 threads)
  #pragma unroll
  for (int t = 0; t < 16; ++t) {
    int idx = tid + t * 256;
    int n = idx >> 7, ch = idx & 127;
    float acc = 0.f;
    if (n < NP3) {
      for (int e = 0; e < NE; ++e) acc += yl[e] * w3[(e * NP3 + n) * CCH + ch];
    } else if (n < NP3 + NP2) {
      int k = n - NP3;
      for (int e = 0; e < NE; ++e) acc += yl[e] * w2[(e * NP2 + k) * CCH + ch];
    } else if (n == NP3 + NP2) {
      for (int e = 0; e < NE; ++e) acc += yl[e] * w1[e * CCH + ch];
    }
    Wm[idx] = acc;
  }
  __syncthreads();

  const int wv = tid >> 5, lane = tid & 31;
  const int h = lane >> 4, ln = lane & 15;
  const int ch = wv * 16 + ln;

  // Per-lane x registers (+ identity and zero slots for pairs/singles/pad).
  float xv[18];
  const float* xp = x + ((size_t)b * CCH + ch) * LL;
  #pragma unroll
  for (int i = 0; i < 16; ++i) xv[i] = xp[i];
  xv[16] = 1.f; xv[17] = 0.f;

  v8f acc0 = {0.f,0.f,0.f,0.f,0.f,0.f,0.f,0.f};
  v8f acc1 = {0.f,0.f,0.f,0.f,0.f,0.f,0.f,0.f};
  const v16h* Abase = (const v16h*)wsA;

  #pragma unroll
  for (int kb = 0; kb < 32; ++kb) {
    // All 32 monomials of this K-block (static indices -> pure VALU).
    float m[32];
    #pragma unroll
    for (int j = 0; j < 32; ++j) {
      const int r = kb * 32 + j;
      m[j] = xv[ROWS.a[r]] * xv[ROWS.b[r]] * xv[ROWS.c[r]];
    }
    // Keep this lane-half's K rows in table order (cndmask, no divergence).
    v16h bf;
    #pragma unroll
    for (int j = 0; j < 16; ++j)
      bf[j] = (_Float16)((j < 8) ? (h ? m[8 + j] : m[j]) : (h ? m[16 + j] : m[8 + j]));

    v16h a0 = Abase[(0 * 32 + kb) * 32 + lane];
    v16h a1 = Abase[(1 * 32 + kb) * 32 + lane];
    acc0 = __builtin_amdgcn_wmma_f32_16x16x32_f16(false, a0, false, bf,
                                                  (short)0, acc0, false, false);
    acc1 = __builtin_amdgcn_wmma_f32_16x16x32_f16(false, a1, false, bf,
                                                  (short)0, acc1, false, false);
  }

  // D layout: lane holds (M = v + 8h, N = ln); M-tile0 -> n=M, tile1 -> n=16+M.
  float part = 0.f;
  #pragma unroll
  for (int v = 0; v < 8; ++v) {
    part += acc0[v] * Wm[(v + 8 * h) * CCH + ch];
    part += acc1[v] * Wm[(16 + v + 8 * h) * CCH + ch];
  }
  part += __shfl_xor(part, 16, 32);   // combine the two n-halves (same ch)
  if (h == 0) out[(size_t)b * CCH + ch] = part;
}

// ---------------------------------------------------------------------------
extern "C" void kernel_launch(void* const* d_in, const int* in_sizes, int n_in,
                              void* d_out, int out_size, void* d_ws, size_t ws_size,
                              hipStream_t stream) {
  const float* x  = (const float*)d_in[0];
  const float* y  = (const float*)d_in[1];
  const float* U3 = (const float*)d_in[2];
  const float* U2 = (const float*)d_in[3];
  const float* U1 = (const float*)d_in[4];
  const float* w3 = (const float*)d_in[5];
  const float* w2 = (const float*)d_in[6];
  const float* w1 = (const float*)d_in[7];
  float*     out  = (float*)d_out;
  _Float16*  wsA  = (_Float16*)d_ws;           // 64 KB swizzled A-operand

  const int Bn = in_sizes[0] / (CCH * LL);     // 5888 nodes

  symcon_prep_A<<<dim3((2 * 32 * 32 * 16) / 256), dim3(256), 0, stream>>>(U3, U2, U1, wsA);
  symcon_main<<<dim3(Bn), dim3(256), 0, stream>>>(x, y, w3, w2, w1, wsA, out);
}
